// CostVolume3Patch_11957188952690
// MI455X (gfx1250) — matrix operand order repaired and checked by reading, not compiled, not run on hardware
//
#include <hip/hip_runtime.h>

typedef _Float16 h8   __attribute__((ext_vector_type(8)));
typedef _Float16 v16h __attribute__((ext_vector_type(16)));
typedef float    v8f  __attribute__((ext_vector_type(8)));

#define BB   2
#define CC   32
#define HH   128
#define WWID 416
#define DD   48
#define CHUNK 64
#define ST_P 304      // halfs per patch row: 288 + 16 pad (keeps 16B alignment, staggers banks)
#define NPL  64       // Pl rows stored
#define NPR  160      // Pr rows stored (halo: [W0-96, W0+64))
#define NSL  112      // Sl rows stored ([W0, W0+112))
#define NSR  160      // Sr rows stored (same halo as Pr)

// ---- WMMA fragment loaders (contiguous k order, same for A and B) ----
// A 16x32 f16: lane L -> row (L&15); halfs K = 16*(L>>4)+[0..15] contiguous
__device__ __forceinline__ v16h frag_a(const _Float16* base, int stride, int row0, int qbase) {
  int lane = threadIdx.x & 31;
  const _Float16* p = base + (size_t)(row0 + (lane & 15)) * stride + qbase + 16 * (lane >> 4);
  h8 lo = *(const h8*)p;
  h8 hi = *(const h8*)(p + 8);
  return __builtin_shufflevector(lo, hi, 0,1,2,3,4,5,6,7,8,9,10,11,12,13,14,15);
}
// B 32x16 f16: lane L -> col (L&15); halfs K = 16*(L>>4)+[0..15] contiguous
__device__ __forceinline__ v16h frag_b(const _Float16* base, int stride, int col0, int qbase) {
  int lane = threadIdx.x & 31;
  const _Float16* p = base + (size_t)(col0 + (lane & 15)) * stride + qbase + 16 * (lane >> 4);
  return *(const v16h*)p;
}

extern "C" __global__ __launch_bounds__(256, 1)
void cost_volume3_wmma(const float* __restrict__ xl, const float* __restrict__ xm,
                       const float* __restrict__ xr, _Float16* __restrict__ out) {
  __shared__ __align__(16) _Float16 s_pl[NPL * ST_P];   //  38,912 B
  __shared__ __align__(16) _Float16 s_pr[NPR * ST_P];   //  97,280 B
  __shared__ __align__(16) _Float16 s_sl[NSL * 32];     //   7,168 B
  __shared__ __align__(16) _Float16 s_sr[NSR * 32];     //  10,240 B
  __shared__ __align__(16) _Float16 s_fl[CHUNK * 32];   //   4,096 B

  const int W0  = blockIdx.x * CHUNK;    // x_l base of this chunk, in [0, 512)
  const int h   = blockIdx.y;
  const int b   = blockIdx.z;
  const int tid = threadIdx.x;

  // ---------------- Phase 1: normalized patches + patch sums ----------------
  // 272 w-tasks: 112 for xl (store patch if <64, always Sl), 160 for xr (patch + Sr).
  // Each task is handled by 4 consecutive lanes (8 channels x 9 taps each).
  const int part = tid & 3;
  for (int task = tid >> 2; task < (NSL + NPR); task += 64) {
    const float* src;
    int w;
    _Float16* prow = nullptr;
    _Float16* srow;
    if (task < NSL) {
      src = xl; w = W0 + task;
      if (task < NPL) prow = &s_pl[task * ST_P];
      srow = &s_sl[task * 32];
    } else {
      int k = task - NSL;
      src = xr; w = W0 - 96 + k;
      prow = &s_pr[k * ST_P];
      srow = &s_sr[k * 32];
    }
    float vals[8][9];
    float ssq = 0.f;
    #pragma unroll
    for (int ci = 0; ci < 8; ++ci) {
      int c = part * 8 + ci;
      const float* basep = src + (size_t)(b * CC + c) * HH * WWID;
      #pragma unroll
      for (int di = 0; di < 3; ++di) {
        int hh2 = h + di - 1;
        #pragma unroll
        for (int dj = 0; dj < 3; ++dj) {
          int ww2 = w + dj - 1;
          float v = 0.f;
          if (hh2 >= 0 && hh2 < HH && ww2 >= 0 && ww2 < WWID)
            v = basep[(size_t)hh2 * WWID + ww2];
          vals[ci][di * 3 + dj] = v;
          ssq += v * v;
        }
      }
    }
    ssq += __shfl_xor(ssq, 1);
    ssq += __shfl_xor(ssq, 2);
    float inv = 1.f / fmaxf(sqrtf(ssq), 1e-3f);   // x / max(||x||, eps)
    #pragma unroll
    for (int ci = 0; ci < 8; ++ci) {
      int c = part * 8 + ci;
      float s9 = 0.f;
      #pragma unroll
      for (int k = 0; k < 9; ++k) {
        s9 += vals[ci][k];
        if (prow) prow[c * 9 + k] = (_Float16)(vals[ci][k] * inv);
      }
      srow[c] = (_Float16)(s9 * inv);
    }
  }

  // ---------------- Phase 2: fl = normalize(xm) over channels ----------------
  if (tid < CHUNK) {
    int w = W0 + tid;
    float v[CC];
    float ssq = 0.f;
    #pragma unroll
    for (int c = 0; c < CC; ++c) {
      float x = 0.f;
      if (w < WWID) x = xm[((size_t)(b * CC + c) * HH + h) * WWID + w];
      v[c] = x;
      ssq += x * x;
    }
    float inv = 1.f / fmaxf(sqrtf(ssq), 1e-3f);
    #pragma unroll
    for (int c = 0; c < CC; ++c) s_fl[tid * 32 + c] = (_Float16)(v[c] * inv);
  }

  __syncthreads();

  // ---------------- Phase 3: banded WMMA GEMMs ----------------
  // 60 tile jobs: [0,28) cLR (4 i-tiles x 7 j-tiles, K=288), [28,44) cL, [44,60) cR (K=32)
  const int wid = tid >> 5;
  const int lane = tid & 31;
  const int n   = lane & 15;
  const int hi4 = lane >> 4;
  const size_t plane  = (size_t)HH * WWID;
  const size_t rowoff = (size_t)h * WWID;
  const float inv_scale = 0.05892556509887897f;   // 1/sqrt(288)

  for (int job = wid; job < 60; job += 8) {
    v8f acc = {};
    if (job < 28) {
      // cLR: M[x_l, x_r] = Pl^T Pr ; out[ch2, d=(x_l-x_r)/2, w=x_l-d]
      int it = job / 7, t = job % 7;
      int i0 = W0 + 16 * it;           // global x_l tile base
      int ia = 16 * it;                // local Pl row
      int jb = 96 + 16 * it - 16 * t;  // local Pr row (x_r - (W0-96))
      #pragma unroll
      for (int kk = 0; kk < 9; ++kk) {
        v16h a  = frag_a(s_pl, ST_P, ia, 32 * kk);
        v16h bm = frag_b(s_pr, ST_P, jb, 32 * kk);
        acc = __builtin_amdgcn_wmma_f32_16x16x32_f16(false, a, false, bm, (short)0, acc, false, false);
      }
      #pragma unroll
      for (int v = 0; v < 8; ++v) {
        int m   = v + 8 * hi4;
        int xL  = i0 + m;
        int xR  = i0 - 16 * t + n;
        int dif = xL - xR;
        if (dif >= 0 && (dif & 1) == 0) {
          int d = dif >> 1;
          int w = xL - d;
          if (d < DD && w >= 0 && w < WWID) {
            float val = fminf(fmaxf(acc[v] * inv_scale, -10.f), 10.f);
            out[((size_t)(b * 3 + 2) * DD + d) * plane + rowoff + w] = (_Float16)val;
          }
        }
      }
    } else if (job < 44) {
      // cL: fl^T Sl ; d = (x - w) in [0,48)
      int jj = job - 28;
      int it = jj >> 2, t = jj & 3;
      int i0 = W0 + 16 * it;
      v16h a  = frag_a(s_fl, 32, 16 * it, 0);
      v16h bm = frag_b(s_sl, 32, 16 * it + 16 * t, 0);
      acc = __builtin_amdgcn_wmma_f32_16x16x32_f16(false, a, false, bm, (short)0, acc, false, false);
      #pragma unroll
      for (int v = 0; v < 8; ++v) {
        int m = v + 8 * hi4;
        int w = i0 + m;
        int d = 16 * t + n - m;
        if (d >= 0 && d < DD && w < WWID) {
          float val = fminf(fmaxf(acc[v] * inv_scale, -10.f), 10.f);
          out[((size_t)(b * 3 + 0) * DD + d) * plane + rowoff + w] = (_Float16)val;
        }
      }
    } else {
      // cR: fl^T Sr ; d = (w - x) in [0,48)
      int jj = job - 44;
      int it = jj >> 2, t = jj & 3;
      int i0 = W0 + 16 * it;
      v16h a  = frag_a(s_fl, 32, 16 * it, 0);
      v16h bm = frag_b(s_sr, 32, 96 + 16 * it - 16 * t, 0);
      acc = __builtin_amdgcn_wmma_f32_16x16x32_f16(false, a, false, bm, (short)0, acc, false, false);
      #pragma unroll
      for (int v = 0; v < 8; ++v) {
        int m = v + 8 * hi4;
        int w = i0 + m;
        int d = 16 * t + m - n;
        if (d >= 0 && d < DD && w < WWID) {
          float val = fminf(fmaxf(acc[v] * inv_scale, -10.f), 10.f);
          out[((size_t)(b * 3 + 1) * DD + d) * plane + rowoff + w] = (_Float16)val;
        }
      }
    }
  }
}

extern "C" void kernel_launch(void* const* d_in, const int* in_sizes, int n_in,
                              void* d_out, int out_size, void* d_ws, size_t ws_size,
                              hipStream_t stream) {
  (void)in_sizes; (void)n_in; (void)out_size; (void)d_ws; (void)ws_size;
  const float* xl = (const float*)d_in[0];
  const float* xm = (const float*)d_in[1];
  const float* xr = (const float*)d_in[2];
  _Float16* out = (_Float16*)d_out;
  // 8 chunks of 64 x_l positions cover x_l = w+d up to 462; every output
  // element (incl. zero-pad regions) is written exactly once.
  dim3 grid(8, HH, BB);
  dim3 block(256);
  hipLaunchKernelGGL(cost_volume3_wmma, grid, block, 0, stream, xl, xm, xr, out);
}
